// Quantizer_72859825209922
// MI455X (gfx1250) — matrix operand order repaired and checked by reading, not compiled, not run on hardware
//
#include <hip/hip_runtime.h>
#include <hip/hip_bf16.h>

typedef __attribute__((ext_vector_type(16))) _Float16 v16h;
typedef __attribute__((ext_vector_type(8)))  _Float16 v8h;
typedef __attribute__((ext_vector_type(8)))  float    v8f;

#define NUM_EMB   8192
#define EMB_DIM   256
#define N_VEC     16384          // 16 * 32 * 32
#define N_ELEM    4194304        // 16 * 256 * 32 * 32
#define J_TILE    16             // codes per LDS tile
#define CODE_HALF 4096           // codes per block (codebook split in 2)
#define ROWS_PER_WAVE 32
#define ROWS_PER_BLOCK 256       // 8 waves * 32 rows

// ---------------------------------------------------------------------------
// Kernel 1: codebook f32 -> f16 + ||e||^2 per row. One wave per row.
// ---------------------------------------------------------------------------
__global__ __launch_bounds__(256) void vq_prep_embed(
    const float* __restrict__ w, _Float16* __restrict__ Eh,
    float* __restrict__ enorm)
{
    const int wave = threadIdx.x >> 5;
    const int lane = threadIdx.x & 31;
    const int r    = blockIdx.x * 8 + wave;
    const float* src = w + (size_t)r * EMB_DIM + lane * 8;
    float4 f0 = *(const float4*)(src);
    float4 f1 = *(const float4*)(src + 4);
    v8h h;
    h[0] = (_Float16)f0.x; h[1] = (_Float16)f0.y;
    h[2] = (_Float16)f0.z; h[3] = (_Float16)f0.w;
    h[4] = (_Float16)f1.x; h[5] = (_Float16)f1.y;
    h[6] = (_Float16)f1.z; h[7] = (_Float16)f1.w;
    *(v8h*)(Eh + (size_t)r * EMB_DIM + lane * 8) = h;
    float ss = f0.x*f0.x + f0.y*f0.y + f0.z*f0.z + f0.w*f0.w
             + f1.x*f1.x + f1.y*f1.y + f1.z*f1.z + f1.w*f1.w;
    #pragma unroll
    for (int off = 16; off >= 1; off >>= 1)
        ss += __shfl_xor(ss, off, 32);
    if (lane == 0) enorm[r] = ss;
}

// ---------------------------------------------------------------------------
// Kernel 2: inputs NCHW f32 -> X[N][C] f16 (row n = b*1024 + h*32 + w).
// ---------------------------------------------------------------------------
__global__ __launch_bounds__(256) void vq_prep_x(
    const float* __restrict__ inp, _Float16* __restrict__ Xh)
{
    const int n = blockIdx.x;
    const int c = threadIdx.x;
    const int b  = n >> 10;
    const int hw = n & 1023;
    float x = inp[(size_t)b * (EMB_DIM * 1024) + (size_t)c * 1024 + hw];
    Xh[(size_t)n * EMB_DIM + c] = (_Float16)x;
}

// ---------------------------------------------------------------------------
// Kernel 3: WMMA GEMM + fused argmin, block-cooperative.
//   block = 8 waves; wave owns 32 A-rows in VGPRs (2 fragment sets).
//   blockIdx -> (rowGroup of 256 rows, codebook half of 4096 codes).
//   Per iteration a 16-code x 256-ch f16 tile (8 KB) is staged into
//   double-buffered LDS via global_load_async_to_lds_b128 (ASYNCcnt),
//   then every wave runs 16 WMMAs (each B fragment reused by 2 A sets).
//   Score = ||e||^2 - 2*x.e  (||x||^2 constant per row, dropped).
// ---------------------------------------------------------------------------
__global__ __launch_bounds__(256) void vq_argmin(
    const _Float16* __restrict__ Xh, const _Float16* __restrict__ Eh,
    const float* __restrict__ enorm,
    float* __restrict__ partMin, int* __restrict__ partIdx)
{
    __shared__ _Float16 tile[2][J_TILE * EMB_DIM];   // 2 x 8 KB

    const int tid  = threadIdx.x;
    const int lane = tid & 31;
    const int wave = tid >> 5;
    const int n = lane & 15;     // column within 16-code tile / A-row in set
    const int g = lane >> 4;     // half-wave group

    const int rowGroup = blockIdx.x >> 1;
    const int half     = blockIdx.x & 1;
    const int rowBase  = rowGroup * ROWS_PER_BLOCK + wave * ROWS_PER_WAVE;
    const int jStart   = half * CODE_HALF;

    // ---- A fragments: 32 rows x 256 ch, resident in VGPRs -----------------
    // 16-bit A 16x32 layout: lane m=lane%16, halves
    // [0..15] = X[m][cbase + 16g .. +15]
    v16h afrag[2][8];
    #pragma unroll
    for (int s = 0; s < 2; ++s) {
        const _Float16* xrow = Xh + (size_t)(rowBase + s * 16 + n) * EMB_DIM;
        #pragma unroll
        for (int kc = 0; kc < 8; ++kc) {
            const _Float16* p = xrow + kc * 32 + g * 16;
            v8h lo = *(const v8h*)(p);
            v8h hi = *(const v8h*)(p + 8);
            afrag[s][kc] = __builtin_shufflevector(lo, hi,
                0,1,2,3,4,5,6,7,8,9,10,11,12,13,14,15);
        }
    }

    float minv[2][8];
    int   mini[2][8];
    #pragma unroll
    for (int s = 0; s < 2; ++s)
        #pragma unroll
        for (int v = 0; v < 8; ++v) { minv[s][v] = 3.0e38f; mini[s][v] = 0; }

    // ---- stage tile 0 (each thread async-copies 32 contiguous bytes) ------
    {
        unsigned lds0 = (unsigned)(uintptr_t)&tile[0][tid * 16];
        unsigned long long ga =
            (unsigned long long)(uintptr_t)(Eh + (size_t)jStart * EMB_DIM + tid * 16);
        asm volatile(
            "global_load_async_to_lds_b128 %0, %2, off\n\t"
            "global_load_async_to_lds_b128 %1, %3, off"
            :: "v"(lds0), "v"(lds0 + 16u), "v"(ga), "v"(ga + 16ull)
            : "memory");
    }
    asm volatile("s_wait_asynccnt 0x0" ::: "memory");
    __syncthreads();

    const int nIter = CODE_HALF / J_TILE;            // 256
    for (int it = 0; it < nIter; ++it) {
        const int buf   = it & 1;
        const int jbase = jStart + it * J_TILE;

        // kick off async DMA of the next tile into the other buffer
        if (it + 1 < nIter) {
            unsigned lds1 = (unsigned)(uintptr_t)&tile[buf ^ 1][tid * 16];
            unsigned long long ga = (unsigned long long)(uintptr_t)
                (Eh + (size_t)(jbase + J_TILE) * EMB_DIM + tid * 16);
            asm volatile(
                "global_load_async_to_lds_b128 %0, %2, off\n\t"
                "global_load_async_to_lds_b128 %1, %3, off"
                :: "v"(lds1), "v"(lds1 + 16u), "v"(ga), "v"(ga + 16ull)
                : "memory");
        }

        // ---- 16 WMMAs from the current LDS tile ---------------------------
        // 16-bit B KxN, contiguous k order (same as A): lane col n, halves
        // [0..15] = E[jbase+n][cbase + 16g .. +15] (32B contiguous in LDS)
        const _Float16* tl = &tile[buf][0] + (size_t)n * EMB_DIM + g * 16;
        v8f accL = {};
        v8f accH = {};
        #pragma unroll
        for (int kc = 0; kc < 8; ++kc) {
            v16h b = *(const v16h*)(tl + kc * 32);
            accL = __builtin_amdgcn_wmma_f32_16x16x32_f16(
                false, afrag[0][kc], false, b, (short)0, accL, false, false);
            accH = __builtin_amdgcn_wmma_f32_16x16x32_f16(
                false, afrag[1][kc], false, b, (short)0, accH, false, false);
        }

        const float en  = enorm[jbase + n];
        const int   jn  = jbase + n;
        // C/D layout: VGPR v holds row (v + 8g), col n for this lane
        #pragma unroll
        for (int v = 0; v < 8; ++v) {
            float s0 = __builtin_fmaf(-2.0f, accL[v], en);
            if (s0 < minv[0][v]) { minv[0][v] = s0; mini[0][v] = jn; }
            float s1 = __builtin_fmaf(-2.0f, accH[v], en);
            if (s1 < minv[1][v]) { minv[1][v] = s1; mini[1][v] = jn; }
        }

        // my async copies done; everyone done reading/writing -> flip buffers
        asm volatile("s_wait_asynccnt 0x0" ::: "memory");
        __syncthreads();
    }

    // ---- reduce across the 16 lanes of each half-wave ---------------------
    #pragma unroll
    for (int off = 8; off >= 1; off >>= 1) {
        #pragma unroll
        for (int s = 0; s < 2; ++s)
            #pragma unroll
            for (int v = 0; v < 8; ++v) {
                float ov = __shfl_xor(minv[s][v], off, 32);
                int   oi = __shfl_xor(mini[s][v], off, 32);
                if (ov < minv[s][v] || (ov == minv[s][v] && oi < mini[s][v])) {
                    minv[s][v] = ov; mini[s][v] = oi;
                }
            }
    }
    if (n == 0) {
        #pragma unroll
        for (int s = 0; s < 2; ++s)
            #pragma unroll
            for (int v = 0; v < 8; ++v) {
                int row = rowBase + s * 16 + v + 8 * g;
                partMin[row * 2 + half] = minv[s][v];
                partIdx[row * 2 + half] = mini[s][v];
            }
    }
}

// ---------------------------------------------------------------------------
// Kernel 3b: combine the two codebook-half candidates per row.
// ---------------------------------------------------------------------------
__global__ __launch_bounds__(256) void vq_combine(
    const float* __restrict__ partMin, const int* __restrict__ partIdx,
    int* __restrict__ bestIdx)
{
    const int r = blockIdx.x * 256 + threadIdx.x;
    float v0 = partMin[2 * r], v1 = partMin[2 * r + 1];
    int   i0 = partIdx[2 * r], i1 = partIdx[2 * r + 1];
    bestIdx[r] = (v1 < v0) ? i1 : i0;   // i0 < i1 always; tie -> lower idx
}

// ---------------------------------------------------------------------------
// Kernel 4: exact f32 gather -> quantized_st output + per-block SSE partials.
// ---------------------------------------------------------------------------
__global__ __launch_bounds__(256) void vq_gather(
    const float* __restrict__ inp, const float* __restrict__ embed,
    const int* __restrict__ bestIdx, float* __restrict__ out,
    float* __restrict__ partial)
{
    __shared__ float red[256];
    const int t = blockIdx.x * 256 + threadIdx.x;
    const int c    = (t >> 10) & 255;
    const int b    = t >> 18;
    const int nrow = (t & 1023) + b * 1024;   // b*1024 + h*32 + w
    const int idx  = bestIdx[nrow];
    const float q = embed[(size_t)idx * EMB_DIM + c];
    const float x = inp[t];
    out[t] = x + (q - x);           // straight-through (matches ref expr)
    const float d = q - x;
    red[threadIdx.x] = d * d;
    __syncthreads();
    #pragma unroll
    for (int s = 128; s > 0; s >>= 1) {
        if (threadIdx.x < s) red[threadIdx.x] += red[threadIdx.x + s];
        __syncthreads();
    }
    if (threadIdx.x == 0) partial[blockIdx.x] = red[0];
}

// ---------------------------------------------------------------------------
// Kernel 5: deterministic final reduction -> both loss scalars.
// ---------------------------------------------------------------------------
__global__ __launch_bounds__(256) void vq_loss(
    const float* __restrict__ partial, int nPartial,
    float* __restrict__ lossOut, float scale)
{
    __shared__ float red[256];
    float s = 0.0f;
    for (int i = threadIdx.x; i < nPartial; i += 256) s += partial[i];
    red[threadIdx.x] = s;
    __syncthreads();
    #pragma unroll
    for (int k = 128; k > 0; k >>= 1) {
        if (threadIdx.x < k) red[threadIdx.x] += red[threadIdx.x + k];
        __syncthreads();
    }
    if (threadIdx.x == 0) {
        float L = red[0] * scale;
        lossOut[0] = L;   // quantization_loss
        lossOut[1] = L;   // commitment_loss (identical value in eval mode)
    }
}

// ---------------------------------------------------------------------------
extern "C" void kernel_launch(void* const* d_in, const int* in_sizes, int n_in,
                              void* d_out, int out_size, void* d_ws, size_t ws_size,
                              hipStream_t stream)
{
    const float* inputs = (const float*)d_in[0];   // [16,256,32,32] f32
    const float* embed  = (const float*)d_in[1];   // [8192,256] f32
    float* out = (float*)d_out;

    char* ws = (char*)d_ws;
    size_t off = 0;
    _Float16* Xh    = (_Float16*)(ws + off); off += (size_t)N_VEC  * EMB_DIM * 2; // 8 MB
    _Float16* Eh    = (_Float16*)(ws + off); off += (size_t)NUM_EMB * EMB_DIM * 2; // 4 MB
    float*    enorm = (float*)   (ws + off); off += (size_t)NUM_EMB * 4;
    int*      bidx  = (int*)     (ws + off); off += (size_t)N_VEC  * 4;
    float*    pmin  = (float*)   (ws + off); off += (size_t)N_VEC  * 2 * 4;
    int*      pidx  = (int*)     (ws + off); off += (size_t)N_VEC  * 2 * 4;
    float*    part  = (float*)   (ws + off); off += (size_t)(N_ELEM / 256) * 4;

    vq_prep_embed<<<NUM_EMB / 8, 256, 0, stream>>>(embed, Eh, enorm);
    vq_prep_x    <<<N_VEC, 256, 0, stream>>>(inputs, Xh);
    vq_argmin    <<<(N_VEC / ROWS_PER_BLOCK) * 2, 256, 0, stream>>>(
                     Xh, Eh, enorm, pmin, pidx);
    vq_combine   <<<N_VEC / 256, 256, 0, stream>>>(pmin, pidx, bidx);
    vq_gather    <<<N_ELEM / 256, 256, 0, stream>>>(inputs, embed, bidx, out, part);
    vq_loss      <<<1, 256, 0, stream>>>(part, N_ELEM / 256,
                                         out + (out_size - 2),
                                         1.0f / (float)N_ELEM);
}
